// Decoder_6622839570644
// MI455X (gfx1250) — compile-verified, matrix operand order checked by reading, not run on hardware
//
#include <hip/hip_runtime.h>

// ---------------------------------------------------------------------------
// Types for CDNA5 WMMA (wave32): v_wmma_f32_16x16x32_bf16
// ---------------------------------------------------------------------------
typedef __bf16 v8bf  __attribute__((ext_vector_type(8)));
typedef __bf16 v16bf __attribute__((ext_vector_type(16)));
typedef float  v8f   __attribute__((ext_vector_type(8)));
typedef int    v4i_  __attribute__((ext_vector_type(4)));

#define B_    128
#define P_    196
#define ENC_  2048
#define D_    1024
#define E_    512
#define AT_   512
#define V_    10000
#define ADIM_ 1024
#define L_    32
#define T_    31

#define VNP_  10112   // V_ rounded up to 128 (W rows padding)
#define VKP_  10016   // V_ rounded up to 32  (K padding)

// ---------------------------------------------------------------------------
// Async global->LDS (CDNA5). Probe-confirmed present on this toolchain; the
// b128 variant takes (v4i addrspace(1)* src, v4i addrspace(3)* dst, imm
// offset, imm cpol).  Fall back to a sync VGPR round-trip if absent.
// ---------------------------------------------------------------------------
#if __has_builtin(__builtin_amdgcn_global_load_async_to_lds_b128)
#define ASYNC_LDS 1
typedef __attribute__((address_space(1))) v4i_* as1_v4i_p;
typedef __attribute__((address_space(3))) v4i_* as3_v4i_p;
#endif

__device__ __forceinline__ void async_b128(const __bf16* g, __bf16* l) {
#if defined(ASYNC_LDS)
  __builtin_amdgcn_global_load_async_to_lds_b128(
      (as1_v4i_p)(unsigned long long)g,
      (as3_v4i_p)(unsigned int)(unsigned long long)l,
      0, 0);
#else
  *(v8bf*)l = *(const v8bf*)g;
#endif
}
__device__ __forceinline__ void wait_async_le3() {
#if defined(ASYNC_LDS)
  asm volatile("s_wait_asynccnt 3" ::: "memory");
#endif
}
__device__ __forceinline__ void wait_async_0() {
#if defined(ASYNC_LDS)
  asm volatile("s_wait_asynccnt 0" ::: "memory");
#endif
}

// ---------------------------------------------------------------------------
// LDS fragment load per ISA 7.12.2 (16-bit A 16x32 layout; B mirrored):
// lane l: row/col = l&15, half = l>>4; two contiguous 16B chunks at
// element offsets half*8 and half*8+16 within a 32-element row.
// ---------------------------------------------------------------------------
__device__ __forceinline__ v16bf frag_ld(const __bf16* rowbase, int half) {
  v8bf lo = *(const v8bf*)(rowbase + half * 8);
  v8bf hi = *(const v8bf*)(rowbase + half * 8 + 16);
  return __builtin_shufflevector(lo, hi, 0,1,2,3,4,5,6,7,8,9,10,11,12,13,14,15);
}

__device__ __forceinline__ float sigf(float x) { return 1.f / (1.f + __expf(-x)); }

__device__ __forceinline__ void storeC(float* C, __bf16* Cbf, int ldcbf,
                                       const float* b1, const float* b2,
                                       int accum, int act,
                                       int row, int col, int N, float v) {
  size_t o = (size_t)row * N + col;
  if (accum) v += C[o];
  if (b1) v += b1[col];
  if (b2) v += b2[col];
  if (act == 1) v = sigf(v);
  C[o] = v;
  if (Cbf) Cbf[(size_t)row * ldcbf + col] = (__bf16)v;
}

// ---------------------------------------------------------------------------
// Tiled bf16 WMMA GEMM: C[M,N] = act(A[M,Kp(lda)] @ W[Np,Kp(ldw)]^T
//                                    + bias1 + bias2 (+ C if accum))
// Requirements (guaranteed by caller): M % 64 == 0, Kp % 32 == 0, operand
// buffers zero-padded to (Np=ceil128(N)) x Kp -> NO guards in the K loop,
// uniform EXEC for WMMA. Block: 256 threads = 8 waves; tile 64(M) x 128(N);
// wave w: row tile (w&3), 64-col half (w>>2), 4 accumulators. Double-buffered
// LDS stages filled by async global->LDS, drained with s_wait_asynccnt.
// ---------------------------------------------------------------------------
__global__ __launch_bounds__(256)
void wmma_gemm_bf16(const __bf16* __restrict__ A, int lda,
                    const __bf16* __restrict__ W, int ldw,
                    const float* __restrict__ bias1,
                    const float* __restrict__ bias2,
                    float* __restrict__ C, __bf16* __restrict__ Cbf, int ldcbf,
                    int M, int N, int Kp, int accum, int act) {
  __shared__ __align__(16) __bf16 As[2][64][32];
  __shared__ __align__(16) __bf16 Ws[2][128][32];

  const int tid  = threadIdx.x;
  const int lane = tid & 31;
  const int wave = tid >> 5;
  const int r    = lane & 15;
  const int half = lane >> 4;
  const int m0   = blockIdx.y * 64;
  const int n0   = blockIdx.x * 128;
  const int tm   = wave & 3;
  const int tn   = (wave >> 2) * 64;

  // Staging: A = 64 rows x 4 chunks (1/thread); W = 128 rows x 4 chunks (2/thread)
  const int sRow = tid >> 2;        // 0..63
  const int sCh  = (tid & 3) * 8;   // 0,8,16,24

  const __bf16* gA  = A + (size_t)(m0 + sRow) * lda + sCh;
  const __bf16* gW0 = W + (size_t)(n0 + sRow) * ldw + sCh;
  const __bf16* gW1 = W + (size_t)(n0 + 64 + sRow) * ldw + sCh;

  v8f acc0 = {}, acc1 = {}, acc2 = {}, acc3 = {};

  const int nk = Kp >> 5;

  // Prologue: fill stage 0
  async_b128(gA, &As[0][sRow][sCh]);
  async_b128(gW0, &Ws[0][sRow][sCh]);
  async_b128(gW1, &Ws[0][64 + sRow][sCh]);

  for (int kc = 0; kc < nk; ++kc) {
    const int s = kc & 1;
    if (kc + 1 < nk) {
      const int ns = s ^ 1;
      const int ko = (kc + 1) << 5;
      async_b128(gA + ko, &As[ns][sRow][sCh]);
      async_b128(gW0 + ko, &Ws[ns][sRow][sCh]);
      async_b128(gW1 + ko, &Ws[ns][64 + sRow][sCh]);
      wait_async_le3();   // drain stage s (older 3), keep stage ns in flight
    } else {
      wait_async_0();
    }
    __syncthreads();

    v16bf a  = frag_ld(&As[s][tm * 16 + r][0], half);
    v16bf b0 = frag_ld(&Ws[s][tn + r][0], half);
    v16bf b1 = frag_ld(&Ws[s][tn + 16 + r][0], half);
    v16bf b2 = frag_ld(&Ws[s][tn + 32 + r][0], half);
    v16bf b3 = frag_ld(&Ws[s][tn + 48 + r][0], half);
    acc0 = __builtin_amdgcn_wmma_f32_16x16x32_bf16(false, a, false, b0, (short)0, acc0, false, false);
    acc1 = __builtin_amdgcn_wmma_f32_16x16x32_bf16(false, a, false, b1, (short)0, acc1, false, false);
    acc2 = __builtin_amdgcn_wmma_f32_16x16x32_bf16(false, a, false, b2, (short)0, acc2, false, false);
    acc3 = __builtin_amdgcn_wmma_f32_16x16x32_bf16(false, a, false, b3, (short)0, acc3, false, false);
    __syncthreads();  // protect stage s from being refilled next-next iter
  }

  // D layout: VGPR i -> (M = i + half*8, N = r) within each 16x16 tile.
  const int col0 = n0 + tn + r;
#pragma unroll
  for (int i = 0; i < 8; ++i) {
    int row = m0 + tm * 16 + half * 8 + i;
    if (row < M) {
      if (col0      < N) storeC(C, Cbf, ldcbf, bias1, bias2, accum, act, row, col0,      N, acc0[i]);
      if (col0 + 16 < N) storeC(C, Cbf, ldcbf, bias1, bias2, accum, act, row, col0 + 16, N, acc1[i]);
      if (col0 + 32 < N) storeC(C, Cbf, ldcbf, bias1, bias2, accum, act, row, col0 + 32, N, acc2[i]);
      if (col0 + 48 < N) storeC(C, Cbf, ldcbf, bias1, bias2, accum, act, row, col0 + 48, N, acc3[i]);
    }
  }
}

// ---------------------------------------------------------------------------
// Small kernels
// ---------------------------------------------------------------------------
__global__ void k_sort(const int* __restrict__ caplen, int* __restrict__ sort_ind,
                       int* __restrict__ dec_len, float* __restrict__ out_dec,
                       float* __restrict__ out_sort) {
  __shared__ int len[B_];
  int i = threadIdx.x;
  len[i] = caplen[i];
  __syncthreads();
  int li = len[i], rank = 0;
  for (int j = 0; j < B_; ++j) {
    int lj = len[j];
    if (lj > li || (lj == li && j < i)) rank++;
  }
  sort_ind[rank] = i;
  __syncthreads();
  int si = sort_ind[i];
  dec_len[i]  = len[si] - 1;
  out_dec[i]  = (float)(len[si] - 1);
  out_sort[i] = (float)si;
}

__global__ void k_caps_out(const int* __restrict__ caps, const int* __restrict__ si,
                           float* __restrict__ out_caps) {
  int idx = blockIdx.x * blockDim.x + threadIdx.x;
  if (idx >= B_ * L_) return;
  int b = idx / L_, t = idx % L_;
  out_caps[idx] = (float)caps[si[b] * L_ + t];
}

__global__ void k_zero(float* __restrict__ p, size_t n) {
  size_t i = (size_t)blockIdx.x * blockDim.x + threadIdx.x;
  size_t s = (size_t)gridDim.x * blockDim.x;
  for (; i < n; i += s) p[i] = 0.f;
}

__global__ void k_zero_bf(__bf16* __restrict__ p, size_t n) {
  size_t i = (size_t)blockIdx.x * blockDim.x + threadIdx.x;
  size_t s = (size_t)gridDim.x * blockDim.x;
  for (; i < n; i += s) p[i] = (__bf16)0.f;
}

__global__ void k_f32_to_bf16(const float* __restrict__ src, __bf16* __restrict__ dst,
                              size_t n) {
  size_t i = (size_t)blockIdx.x * blockDim.x + threadIdx.x;
  size_t s = (size_t)gridDim.x * blockDim.x;
  for (; i < n; i += s) dst[i] = (__bf16)src[i];
}

// Strided convert with zero padding: dst[Np x Kp] from src[N x K]
__global__ void k_cvt_pad(const float* __restrict__ src, __bf16* __restrict__ dst,
                          int N, int K, int Np, int Kp) {
  size_t n = (size_t)Np * Kp;
  size_t i = (size_t)blockIdx.x * blockDim.x + threadIdx.x;
  size_t s = (size_t)gridDim.x * blockDim.x;
  for (; i < n; i += s) {
    int rr = (int)(i / Kp), cc = (int)(i % Kp);
    dst[i] = (rr < N && cc < K) ? (__bf16)src[(size_t)rr * K + cc] : (__bf16)0.f;
  }
}

__global__ void k_gather_rows_bf(const float* __restrict__ src, const int* __restrict__ idx,
                                 __bf16* __restrict__ dst, int rows, int cols) {
  int i = blockIdx.x * blockDim.x + threadIdx.x;
  if (i >= rows * cols) return;
  int r = i / cols, c = i % cols;
  dst[i] = (__bf16)src[(size_t)idx[r] * cols + c];
}

__global__ void k_gather_emb(const float* __restrict__ emb, const int* __restrict__ caps,
                             const int* __restrict__ si, __bf16* __restrict__ embs_bf) {
  int idx = blockIdx.x * blockDim.x + threadIdx.x;
  if (idx >= B_ * T_ * E_) return;
  int b = idx / (T_ * E_);
  int r = idx % (T_ * E_);
  int t = r / E_, e = r % E_;
  int tok = caps[si[b] * L_ + t];
  embs_bf[idx] = (__bf16)emb[(size_t)tok * E_ + e];
}

__global__ void k_mean_bf(const float* __restrict__ enc, __bf16* __restrict__ mean_bf) {
  int idx = blockIdx.x * blockDim.x + threadIdx.x;
  if (idx >= B_ * ENC_) return;
  int b = idx / ENC_, e = idx % ENC_;
  const float* base = enc + (size_t)b * P_ * ENC_ + e;
  float s = 0.f;
  for (int p = 0; p < P_; ++p) s += base[(size_t)p * ENC_];
  mean_bf[idx] = (__bf16)(s * (1.0f / (float)P_));
}

// LSTM gate nonlinearity + masked state update. t < 0 => unconditional.
__global__ void k_lstm_apply(const float* __restrict__ g, float* __restrict__ h,
                             float* __restrict__ c, __bf16* __restrict__ hbf,
                             int Dd, const int* __restrict__ dec_len, int t) {
  int idx = blockIdx.x * blockDim.x + threadIdx.x;
  if (idx >= B_ * Dd) return;
  int b = idx / Dd, d = idx % Dd;
  size_t base = (size_t)b * 4 * Dd;
  float gi = g[base + d];
  float gf = g[base + Dd + d];
  float gg = g[base + 2 * Dd + d];
  float go = g[base + 3 * Dd + d];
  float c2 = sigf(gf) * c[idx] + sigf(gi) * tanhf(gg);
  float h2 = sigf(go) * tanhf(c2);
  bool active = (t < 0) || (t < dec_len[b]);
  float hn = active ? h2 : h[idx];
  if (active) { h[idx] = h2; c[idx] = c2; }
  hbf[idx] = (__bf16)hn;
}

// Attention: scores = relu(att1[b,p,:] + att2[b,:]) . W_fa + b_fa,
// softmax over p, awe[b,:] = gate * sum_p alpha_p * enc[b,p,:]
__global__ __launch_bounds__(256)
void k_attention(const float* __restrict__ att1, const float* __restrict__ att2,
                 const float* __restrict__ Wfa, const float* __restrict__ bfa,
                 const float* __restrict__ enc, const float* __restrict__ gate,
                 float* __restrict__ awe) {
  __shared__ float sc[224];
  int b    = blockIdx.x;
  int tid  = threadIdx.x;
  int lane = tid & 31;
  int w    = tid >> 5;
  const float* a2 = att2 + (size_t)b * AT_;

  for (int p = w; p < P_; p += 8) {
    const float* a1 = att1 + ((size_t)b * P_ + p) * AT_;
    float acc = 0.f;
    for (int k = lane; k < AT_; k += 32) {
      float v = a1[k] + a2[k];
      v = v > 0.f ? v : 0.f;
      acc += v * Wfa[k];
    }
    for (int off = 16; off > 0; off >>= 1) acc += __shfl_down(acc, off, 32);
    if (lane == 0) sc[p] = acc + bfa[0];
  }
  __syncthreads();

  if (w == 0) {
    float m = -1e30f;
    for (int p = lane; p < P_; p += 32) m = fmaxf(m, sc[p]);
    for (int off = 16; off > 0; off >>= 1) m = fmaxf(m, __shfl_down(m, off, 32));
    m = __shfl(m, 0, 32);
    float s = 0.f;
    for (int p = lane; p < P_; p += 32) { float e = __expf(sc[p] - m); sc[p] = e; s += e; }
    for (int off = 16; off > 0; off >>= 1) s += __shfl_down(s, off, 32);
    s = __shfl(s, 0, 32);
    float inv = 1.f / s;
    for (int p = lane; p < P_; p += 32) sc[p] *= inv;
  }
  __syncthreads();

  const float* eb = enc + (size_t)b * P_ * ENC_;
  for (int e = tid; e < ENC_; e += 256) {
    float acc = 0.f;
    for (int p = 0; p < P_; ++p) acc += eb[(size_t)p * ENC_ + e] * sc[p];
    awe[(size_t)b * ENC_ + e] = acc * gate[(size_t)b * ENC_ + e];
  }
}

// xin[b] = [x_t[b] (bf16, strided) | bf16(awe[b])]
__global__ void k_build_xin(const __bf16* __restrict__ xt, int ldx,
                            const float* __restrict__ awe, __bf16* __restrict__ xin) {
  int idx = blockIdx.x * blockDim.x + threadIdx.x;
  const int CW = E_ + ENC_;
  if (idx >= B_ * CW) return;
  int b = idx / CW, j = idx % CW;
  xin[idx] = (j < E_) ? xt[(size_t)b * ldx + j]
                      : (__bf16)awe[(size_t)b * ENC_ + (j - E_)];
}

__global__ void k_write_preds(const float* __restrict__ ptmp,
                              const int* __restrict__ dec_len, int t,
                              float* __restrict__ out_pred) {
  int idx = blockIdx.x * blockDim.x + threadIdx.x;
  if (idx >= B_ * V_) return;
  int b = idx / V_, v = idx % V_;
  bool active = t < dec_len[b];
  out_pred[((size_t)b * T_ + t) * V_ + v] = active ? ptmp[idx] : 0.f;
}

// ---------------------------------------------------------------------------
// Host side
// ---------------------------------------------------------------------------
extern "C" void kernel_launch(void* const* d_in, const int* in_sizes, int n_in,
                              void* d_out, int out_size, void* d_ws, size_t ws_size,
                              hipStream_t stream) {
  (void)in_sizes; (void)n_in; (void)out_size; (void)ws_size;

  const float* attrs  = (const float*)d_in[0];
  const float* enc    = (const float*)d_in[1];
  const int*   caps   = (const int*)d_in[2];
  const int*   caplen = (const int*)d_in[3];
  const float* emb    = (const float*)d_in[4];
  const float* W_x0  = (const float*)d_in[5];  const float* b_x0  = (const float*)d_in[6];
  const float* W_h1  = (const float*)d_in[7];  const float* b_h1  = (const float*)d_in[8];
  const float* W_c1  = (const float*)d_in[9];  const float* b_c1  = (const float*)d_in[10];
  const float* W_fc1 = (const float*)d_in[11]; const float* b_fc1 = (const float*)d_in[12];
  const float* W_ea  = (const float*)d_in[13]; const float* b_ea  = (const float*)d_in[14];
  const float* W_da  = (const float*)d_in[15]; const float* b_da  = (const float*)d_in[16];
  const float* W_fa  = (const float*)d_in[17]; const float* b_fa  = (const float*)d_in[18];
  const float* W_h2  = (const float*)d_in[19]; const float* b_h2  = (const float*)d_in[20];
  const float* W_c2  = (const float*)d_in[21]; const float* b_c2  = (const float*)d_in[22];
  const float* W_fb  = (const float*)d_in[23]; const float* b_fb  = (const float*)d_in[24];
  const float* W_fc2 = (const float*)d_in[25]; const float* b_fc2 = (const float*)d_in[26];
  const float* W_pre = (const float*)d_in[27]; const float* b_pre = (const float*)d_in[28];
  const float* W_ih1 = (const float*)d_in[29]; const float* b_ih1 = (const float*)d_in[30];
  const float* W_hh1 = (const float*)d_in[31]; const float* b_hh1 = (const float*)d_in[32];
  const float* W_ih2 = (const float*)d_in[33]; const float* b_ih2 = (const float*)d_in[34];
  const float* W_hh2 = (const float*)d_in[35]; const float* b_hh2 = (const float*)d_in[36];

  // Output layout (flat, return order): predictions, caps_s, dec_len, alphas, sort_ind
  float* out      = (float*)d_out;
  float* out_pred = out;
  float* out_caps = out_pred + (size_t)B_ * T_ * V_;
  float* out_dec  = out_caps + (size_t)B_ * L_;
  float* out_alph = out_dec + B_;
  float* out_sort = out_alph + (size_t)B_ * T_ * P_;

  // Workspace bump allocator (256B aligned)
  char* wsb = (char*)d_ws;
  size_t off = 0;
  auto alloc = [&](size_t bytes) -> void* {
    void* p = wsb + off;
    off = (off + bytes + 255) & ~(size_t)255;
    return p;
  };

  int*    sort_ind_i = (int*)alloc(B_ * sizeof(int));
  int*    dec_len_i  = (int*)alloc(B_ * sizeof(int));
  __bf16* attrs_bf   = (__bf16*)alloc((size_t)B_ * ADIM_ * 2);
  __bf16* attrs_s_bf = (__bf16*)alloc((size_t)B_ * ADIM_ * 2);
  __bf16* mean_bf    = (__bf16*)alloc((size_t)B_ * ENC_ * 2);
  __bf16* enc_bf     = (__bf16*)alloc((size_t)B_ * P_ * ENC_ * 2);
  float*  att1       = (float*)alloc((size_t)B_ * P_ * AT_ * 4);
  __bf16* embs_bf    = (__bf16*)alloc((size_t)B_ * T_ * E_ * 2);

  __bf16* Wx0b  = (__bf16*)alloc((size_t)E_ * ADIM_ * 2);
  __bf16* Wh1b  = (__bf16*)alloc((size_t)D_ * ADIM_ * 2);
  __bf16* Wc1b  = (__bf16*)alloc((size_t)D_ * ADIM_ * 2);
  __bf16* Wfc1b = (__bf16*)alloc((size_t)VNP_ * D_ * 2);      // padded rows
  __bf16* Weab  = (__bf16*)alloc((size_t)AT_ * ENC_ * 2);
  __bf16* Wdab  = (__bf16*)alloc((size_t)AT_ * D_ * 2);
  __bf16* Wh2b  = (__bf16*)alloc((size_t)D_ * ENC_ * 2);
  __bf16* Wc2b  = (__bf16*)alloc((size_t)D_ * ENC_ * 2);
  __bf16* Wfbb  = (__bf16*)alloc((size_t)ENC_ * D_ * 2);
  __bf16* Wfc2b = (__bf16*)alloc((size_t)VNP_ * D_ * 2);      // padded rows
  __bf16* Wpreb = (__bf16*)alloc((size_t)VNP_ * VKP_ * 2);    // padded rows+K
  __bf16* Wih1b = (__bf16*)alloc((size_t)4 * D_ * E_ * 2);
  __bf16* Whh1b = (__bf16*)alloc((size_t)4 * D_ * D_ * 2);
  __bf16* Wih2b = (__bf16*)alloc((size_t)4 * D_ * (E_ + ENC_) * 2);
  __bf16* Whh2b = (__bf16*)alloc((size_t)4 * D_ * D_ * 2);

  float*  x0f  = (float*)alloc((size_t)B_ * E_ * 4);
  __bf16* x0bf = (__bf16*)alloc((size_t)B_ * E_ * 2);
  float*  h1   = (float*)alloc((size_t)B_ * D_ * 4);
  float*  c1   = (float*)alloc((size_t)B_ * D_ * 4);
  float*  h2   = (float*)alloc((size_t)B_ * D_ * 4);
  float*  c2   = (float*)alloc((size_t)B_ * D_ * 4);
  __bf16* h1bf = (__bf16*)alloc((size_t)B_ * D_ * 2);
  __bf16* h2bf = (__bf16*)alloc((size_t)B_ * D_ * 2);
  float*  g1   = (float*)alloc((size_t)B_ * 4 * D_ * 4);
  float*  g2   = (float*)alloc((size_t)B_ * 4 * D_ * 4);
  float*  att2 = (float*)alloc((size_t)B_ * AT_ * 4);
  float*  gateb= (float*)alloc((size_t)B_ * ENC_ * 4);
  float*  awe  = (float*)alloc((size_t)B_ * ENC_ * 4);
  __bf16* xinb = (__bf16*)alloc((size_t)B_ * (E_ + ENC_) * 2);
  float*  sbuf = (float*)alloc((size_t)B_ * V_ * 4);
  __bf16* sbf  = (__bf16*)alloc((size_t)B_ * VKP_ * 2);       // padded K for pre-GEMM
  float*  ptmp = (float*)alloc((size_t)B_ * V_ * 4);

  auto cvt = [&](const float* src, __bf16* dst, size_t n) {
    size_t blocks = (n + 255) / 256; if (blocks > 4096) blocks = 4096;
    k_f32_to_bf16<<<(unsigned)blocks, 256, 0, stream>>>(src, dst, n);
  };
  auto cvtpad = [&](const float* src, __bf16* dst, int N, int K, int Np, int Kp) {
    size_t n = (size_t)Np * Kp;
    size_t blocks = (n + 255) / 256; if (blocks > 4096) blocks = 4096;
    k_cvt_pad<<<(unsigned)blocks, 256, 0, stream>>>(src, dst, N, K, Np, Kp);
  };
  auto gemm = [&](const __bf16* A, int lda, const __bf16* W, int ldw,
                  const float* bias1, const float* bias2,
                  float* C, __bf16* Cbf, int ldcbf,
                  int M, int N, int Kp, int accum, int act) {
    dim3 g((N + 127) / 128, (M + 63) / 64);
    wmma_gemm_bf16<<<g, 256, 0, stream>>>(A, lda, W, ldw, bias1, bias2,
                                          C, Cbf, ldcbf, M, N, Kp, accum, act);
  };

  // ---- Setup ----
  k_sort<<<1, B_, 0, stream>>>(caplen, sort_ind_i, dec_len_i, out_dec, out_sort);
  k_caps_out<<<(B_ * L_ + 255) / 256, 256, 0, stream>>>(caps, sort_ind_i, out_caps);
  k_zero<<<1024, 256, 0, stream>>>(out_alph, (size_t)B_ * T_ * P_);
  k_zero_bf<<<1024, 256, 0, stream>>>(sbf, (size_t)B_ * VKP_);  // K-pad stays 0

  cvt(attrs, attrs_bf, (size_t)B_ * ADIM_);
  k_gather_rows_bf<<<(B_ * ADIM_ + 255) / 256, 256, 0, stream>>>(attrs, sort_ind_i,
                                                                 attrs_s_bf, B_, ADIM_);
  k_gather_emb<<<(B_ * T_ * E_ + 255) / 256, 256, 0, stream>>>(emb, caps, sort_ind_i,
                                                               embs_bf);
  k_mean_bf<<<(B_ * ENC_ + 255) / 256, 256, 0, stream>>>(enc, mean_bf);
  cvt(enc, enc_bf, (size_t)B_ * P_ * ENC_);

  cvt(W_x0, Wx0b, (size_t)E_ * ADIM_);
  cvt(W_h1, Wh1b, (size_t)D_ * ADIM_);
  cvt(W_c1, Wc1b, (size_t)D_ * ADIM_);
  cvt(W_ea, Weab, (size_t)AT_ * ENC_);
  cvt(W_da, Wdab, (size_t)AT_ * D_);
  cvt(W_h2, Wh2b, (size_t)D_ * ENC_);
  cvt(W_c2, Wc2b, (size_t)D_ * ENC_);
  cvt(W_fb, Wfbb, (size_t)ENC_ * D_);
  cvt(W_ih1, Wih1b, (size_t)4 * D_ * E_);
  cvt(W_hh1, Whh1b, (size_t)4 * D_ * D_);
  cvt(W_ih2, Wih2b, (size_t)4 * D_ * (E_ + ENC_));
  cvt(W_hh2, Whh2b, (size_t)4 * D_ * D_);
  cvtpad(W_fc1, Wfc1b, V_, D_, VNP_, D_);
  cvtpad(W_fc2, Wfc2b, V_, D_, VNP_, D_);
  cvtpad(W_pre, Wpreb, V_, V_, VNP_, VKP_);

  // Initial projections (x0 uses UNSORTED attrs; h1/c1 use sorted — as in ref)
  gemm(attrs_bf,   ADIM_, Wx0b, ADIM_, b_x0, nullptr, x0f, x0bf, E_, B_, E_, ADIM_, 0, 0);
  gemm(attrs_s_bf, ADIM_, Wh1b, ADIM_, b_h1, nullptr, h1,  h1bf, D_, B_, D_, ADIM_, 0, 0);
  gemm(attrs_s_bf, ADIM_, Wc1b, ADIM_, b_c1, nullptr, c1,  nullptr, 0, B_, D_, ADIM_, 0, 0);
  gemm(mean_bf,    ENC_,  Wh2b, ENC_,  b_h2, nullptr, h2,  h2bf, D_, B_, D_, ENC_, 0, 0);
  gemm(mean_bf,    ENC_,  Wc2b, ENC_,  b_c2, nullptr, c2,  nullptr, 0, B_, D_, ENC_, 0, 0);

  // att1 = encoder_out @ W_ea^T + b_ea   (M = B*P = 25088, multiple of 64)
  gemm(enc_bf, ENC_, Weab, ENC_, b_ea, nullptr, att1, nullptr, 0,
       B_ * P_, AT_, ENC_, 0, 0);

  // Initial (unmasked) LSTM1 step with x0
  gemm(x0bf, E_, Wih1b, E_, b_ih1, b_hh1, g1, nullptr, 0, B_, 4 * D_, E_, 0, 0);
  gemm(h1bf, D_, Whh1b, D_, nullptr, nullptr, g1, nullptr, 0, B_, 4 * D_, D_, 1, 0);
  k_lstm_apply<<<(B_ * D_ + 255) / 256, 256, 0, stream>>>(g1, h1, c1, h1bf, D_,
                                                          dec_len_i, -1);

  // ---- Time loop ----
  for (int t = 0; t < T_; ++t) {
    const __bf16* xt = embs_bf + (size_t)t * E_;  // row stride T_*E_

    // LSTM1
    gemm(xt, T_ * E_, Wih1b, E_, b_ih1, b_hh1, g1, nullptr, 0, B_, 4 * D_, E_, 0, 0);
    gemm(h1bf, D_, Whh1b, D_, nullptr, nullptr, g1, nullptr, 0, B_, 4 * D_, D_, 1, 0);
    k_lstm_apply<<<(B_ * D_ + 255) / 256, 256, 0, stream>>>(g1, h1, c1, h1bf, D_,
                                                            dec_len_i, t);

    // Attention (uses pre-update h2)
    gemm(h2bf, D_, Wdab, D_, b_da, nullptr, att2, nullptr, 0, B_, AT_, D_, 0, 0);
    gemm(h2bf, D_, Wfbb, D_, b_fb, nullptr, gateb, nullptr, 0, B_, ENC_, D_, 0, 1);
    k_attention<<<B_, 256, 0, stream>>>(att1, att2, W_fa, b_fa, enc, gateb, awe);

    // LSTM2 on [x_t | gated awe]
    k_build_xin<<<(B_ * (E_ + ENC_) + 255) / 256, 256, 0, stream>>>(xt, T_ * E_,
                                                                    awe, xinb);
    gemm(xinb, E_ + ENC_, Wih2b, E_ + ENC_, b_ih2, b_hh2, g2, nullptr, 0,
         B_, 4 * D_, E_ + ENC_, 0, 0);
    gemm(h2bf, D_, Whh2b, D_, nullptr, nullptr, g2, nullptr, 0, B_, 4 * D_, D_, 1, 0);
    k_lstm_apply<<<(B_ * D_ + 255) / 256, 256, 0, stream>>>(g2, h2, c2, h2bf, D_,
                                                            dec_len_i, t);

    // preds = (h1@Wfc1^T + b_fc1 + h2@Wfc2^T + b_fc2) @ W_pre^T + b_pre
    gemm(h1bf, D_, Wfc1b, D_, b_fc1, nullptr, sbuf, nullptr, 0, B_, V_, D_, 0, 0);
    gemm(h2bf, D_, Wfc2b, D_, b_fc2, nullptr, sbuf, sbf, VKP_, B_, V_, D_, 1, 0);
    gemm(sbf, VKP_, Wpreb, VKP_, b_pre, nullptr, ptmp, nullptr, 0,
         B_, V_, VKP_, 0, 0);
    k_write_preds<<<(B_ * V_ + 255) / 256, 256, 0, stream>>>(ptmp, dec_len_i, t,
                                                             out_pred);
  }
}
